// Qwen2MoeSparseMoeBlock_4054449127762
// MI455X (gfx1250) — compile-verified, matrix operand order checked by reading, not run on hardware
//
#include <hip/hip_runtime.h>
#include <hip/hip_bf16.h>

typedef __attribute__((ext_vector_type(16))) __bf16 v16bf;
typedef __attribute__((ext_vector_type(8)))  float  v8f;

#define HDIM  2048
#define NEXP  16
#define IDIM  1408
#define ISDIM 5632
#define NTOK  2048

// ---------- helpers ----------

// round-to-nearest-even f32 -> bf16, result left in bits [31:16]
__device__ __forceinline__ unsigned rne_hi(float f) {
  unsigned u = __float_as_uint(f);
  return u + 0x7FFFu + ((u >> 16) & 1u);
}

// pack two bf16 (lo -> bits[15:0], hi -> bits[31:16]) with one v_perm_b32
__device__ __forceinline__ unsigned pk2bf(float lo, float hi) {
  // D.byte0 = ra.byte2, D.byte1 = ra.byte3 (codes 2,3 select S1 bytes)
  // D.byte2 = rb.byte2, D.byte3 = rb.byte3 (codes 6,7 select S0 bytes)
  return __builtin_amdgcn_perm(rne_hi(hi), rne_hi(lo), 0x07060302u);
}

// scalar: hi-half store pattern -> ds_store_b16_d16_hi
__device__ __forceinline__ unsigned short f2bf(float f) {
  return (unsigned short)(rne_hi(f) >> 16);
}

__device__ __forceinline__ float fast_sigmoid(float x) {
  return __builtin_amdgcn_rcpf(1.0f + __expf(-x));   // v_rcp_f32, not full div
}

// Load one 16x32 bf16 WMMA fragment from an LDS tile stored row-major with
// row stride 32 (bf16 elems). CDNA5 16-bit A/B frag layout: lane%16 = M/N row,
// lane>=16 carries K halves 8..15 / 24..31; VGPR0..3 = 8 contiguous K elems,
// VGPR4..7 = the +16 K block -> two b128 LDS reads per lane.
__device__ __forceinline__ v16bf load_frag(const unsigned short* base, int rowbase, int lane) {
  const int m  = lane & 15;
  const int kb = (lane >> 4) << 3;  // 0 or 8
  union { uint4 u[2]; v16bf v; } f;
  f.u[0] = *(const uint4*)(base + (rowbase + m) * 32 + kb);
  f.u[1] = *(const uint4*)(base + (rowbase + m) * 32 + kb + 16);
  return f.v;
}

// ---------- router: logits -> softmax -> top4 combine weights + shared gate ----------

__global__ __launch_bounds__(256) void router_kernel(
    const float* __restrict__ X, const float* __restrict__ Wr,
    const float* __restrict__ Wsg,
    float* __restrict__ combine, float* __restrict__ sg, float* __restrict__ out)
{
  __shared__ float red[256];
  __shared__ float red2[256];
  __shared__ float logits[NEXP];
  const int t = blockIdx.x;
  const int tid = threadIdx.x;
  const float* x = X + (size_t)t * HDIM;

  const int e = tid & 15;
  const int g = tid >> 4;            // 16 threads per expert
  float p = 0.f;
  for (int j = 0; j < HDIM / 16; ++j) {
    int h = g + j * 16;
    p += x[h] * Wr[h * NEXP + e];
  }
  red[tid] = p;

  float q = 0.f;
  for (int j = 0; j < HDIM / 256; ++j) {
    int h = tid + j * 256;
    q += x[h] * Wsg[h];
  }
  red2[tid] = q;
  __syncthreads();

  if (tid < NEXP) {
    float s = 0.f;
    for (int j = 0; j < 16; ++j) s += red[j * 16 + tid];
    logits[tid] = s;
  }
  __syncthreads();

  if (tid == 0) {
    float mx = logits[0];
    for (int i = 1; i < NEXP; ++i) mx = fmaxf(mx, logits[i]);
    float pe[NEXP]; float sum = 0.f;
    for (int i = 0; i < NEXP; ++i) { pe[i] = __expf(logits[i] - mx); sum += pe[i]; }
    float inv = __builtin_amdgcn_rcpf(sum);
    float cw[NEXP];
    bool used[NEXP];
    for (int i = 0; i < NEXP; ++i) { pe[i] *= inv; cw[i] = 0.f; used[i] = false; }
    for (int k = 0; k < 4; ++k) {       // top-4, no renorm (norm_topk_prob=False)
      int best = 0; float bv = -1.f;
      for (int i = 0; i < NEXP; ++i)
        if (!used[i] && pe[i] > bv) { bv = pe[i]; best = i; }
      used[best] = true;
      cw[best] = pe[best];
    }
    for (int i = 0; i < NEXP; ++i) combine[(size_t)t * NEXP + i] = cw[i];
    float st = 0.f;
    for (int j = 0; j < 256; ++j) st += red2[j];
    sg[t] = fast_sigmoid(st);
  }

  // zero-init this token's output row so the down-GEMMs can accumulate
  for (int j = 0; j < HDIM / 256; ++j)
    out[(size_t)t * HDIM + tid + j * 256] = 0.f;
}

// ---------- fused gate/up GEMM: A = silu(X Wg) * (X Wu) * rowscale ----------
// BM=128, BN=64, BK=32. 256 threads = 8 waves (4x2), each wave a 32x32 subtile
// per matrix -> 8 v_wmma_f32_16x16x32_bf16 per K-step per wave.
// Double-buffered LDS; next tile's global loads issued before current compute.

__global__ __launch_bounds__(256) void gateup_kernel(
    const float* __restrict__ X,
    const float* __restrict__ Wg,
    const float* __restrict__ Wu,
    const float* __restrict__ rowscale, int rs_stride,
    float* __restrict__ Aout, int N)
{
  __shared__ __align__(16) unsigned short Al [2][128 * 32];
  __shared__ __align__(16) unsigned short Bgl[2][ 64 * 32];
  __shared__ __align__(16) unsigned short Bul[2][ 64 * 32];

  const int tid  = threadIdx.x;
  const int lane = tid & 31;
  const int wid  = tid >> 5;
  const int wm   = wid >> 1;   // 0..3
  const int wn   = wid & 1;    // 0..1
  const int t0   = blockIdx.y * 128;
  const int n0   = blockIdx.x * 64;

  v8f ag[2][2] = {};
  v8f au[2][2] = {};

  float4 xr[4], gr[2], ur[2];

  auto gload = [&](int k0) {
    #pragma unroll
    for (int it = 0; it < 4; ++it) {
      int idx = it * 256 + tid;
      int row = idx >> 3;
      int c4  = (idx & 7) << 2;
      xr[it] = *(const float4*)(X + (size_t)(t0 + row) * HDIM + k0 + c4);
    }
    #pragma unroll
    for (int it = 0; it < 2; ++it) {
      int idx = it * 256 + tid;
      int kk  = idx >> 4;          // 0..31
      int c4  = (idx & 15) << 2;   // n within tile
      size_t go = (size_t)(k0 + kk) * N + n0 + c4;
      gr[it] = *(const float4*)(Wg + go);
      ur[it] = *(const float4*)(Wu + go);
    }
  };

  auto lstore = [&](int b) {
    #pragma unroll
    for (int it = 0; it < 4; ++it) {
      int idx = it * 256 + tid;
      int row = idx >> 3;
      int c4  = (idx & 7) << 2;
      unsigned* dst = (unsigned*)(&Al[b][row * 32 + c4]);
      dst[0] = pk2bf(xr[it].x, xr[it].y);
      dst[1] = pk2bf(xr[it].z, xr[it].w);
    }
    #pragma unroll
    for (int it = 0; it < 2; ++it) {
      int idx = it * 256 + tid;
      int kk  = idx >> 4;
      int c4  = (idx & 15) << 2;
      Bgl[b][(c4 + 0) * 32 + kk] = f2bf(gr[it].x);
      Bgl[b][(c4 + 1) * 32 + kk] = f2bf(gr[it].y);
      Bgl[b][(c4 + 2) * 32 + kk] = f2bf(gr[it].z);
      Bgl[b][(c4 + 3) * 32 + kk] = f2bf(gr[it].w);
      Bul[b][(c4 + 0) * 32 + kk] = f2bf(ur[it].x);
      Bul[b][(c4 + 1) * 32 + kk] = f2bf(ur[it].y);
      Bul[b][(c4 + 2) * 32 + kk] = f2bf(ur[it].z);
      Bul[b][(c4 + 3) * 32 + kk] = f2bf(ur[it].w);
    }
  };

  const int NK = HDIM / 32;
  gload(0);
  lstore(0);
  __syncthreads();

  #pragma unroll 1
  for (int ks = 0; ks < NK; ++ks) {
    const int cur = ks & 1;
    if (ks + 1 < NK) gload((ks + 1) * 32);   // overlap next tile with compute

    v16bf af[2], bgf[2], buf2[2];
    #pragma unroll
    for (int mt = 0; mt < 2; ++mt) af[mt] = load_frag(Al[cur], wm * 32 + mt * 16, lane);
    #pragma unroll
    for (int nt = 0; nt < 2; ++nt) {
      bgf[nt]  = load_frag(Bgl[cur], wn * 32 + nt * 16, lane);
      buf2[nt] = load_frag(Bul[cur], wn * 32 + nt * 16, lane);
    }
    #pragma unroll
    for (int mt = 0; mt < 2; ++mt)
      #pragma unroll
      for (int nt = 0; nt < 2; ++nt) {
        ag[mt][nt] = __builtin_amdgcn_wmma_f32_16x16x32_bf16(
            false, af[mt], false, bgf[nt], (short)0, ag[mt][nt], false, false);
        au[mt][nt] = __builtin_amdgcn_wmma_f32_16x16x32_bf16(
            false, af[mt], false, buf2[nt], (short)0, au[mt][nt], false, false);
      }

    if (ks + 1 < NK) {
      lstore(cur ^ 1);
      __syncthreads();
    }
  }

  // epilogue: silu(g)*u*rowscale  (C/D layout: lane%16 = N, lanes>=16 -> M+8, vgpr r -> M)
  const int rl = (lane >> 4) << 3;
  const int cl = lane & 15;
  #pragma unroll
  for (int mt = 0; mt < 2; ++mt)
    #pragma unroll
    for (int nt = 0; nt < 2; ++nt)
      #pragma unroll
      for (int r = 0; r < 8; ++r) {
        int row = t0 + wm * 32 + mt * 16 + rl + r;
        int col = n0 + wn * 32 + nt * 16 + cl;
        float gv = ag[mt][nt][r];
        float uv = au[mt][nt][r];
        float sv = gv * fast_sigmoid(gv);
        float rs = rowscale[(size_t)row * rs_stride];
        Aout[(size_t)row * N + col] = sv * uv * rs;
      }
}

// ---------- down GEMM: Out[T,H] += A[T,K] @ W[K,H] ----------

__global__ __launch_bounds__(256) void down_kernel(
    const float* __restrict__ Ain, const float* __restrict__ W,
    float* __restrict__ Out, int K)
{
  __shared__ __align__(16) unsigned short Al[2][128 * 32];
  __shared__ __align__(16) unsigned short Bl[2][ 64 * 32];

  const int tid  = threadIdx.x;
  const int lane = tid & 31;
  const int wid  = tid >> 5;
  const int wm   = wid >> 1;
  const int wn   = wid & 1;
  const int t0   = blockIdx.y * 128;
  const int n0   = blockIdx.x * 64;

  v8f acc[2][2] = {};

  float4 ar[4], wr[2];

  auto gload = [&](int k0) {
    #pragma unroll
    for (int it = 0; it < 4; ++it) {
      int idx = it * 256 + tid;
      int row = idx >> 3;
      int c4  = (idx & 7) << 2;
      ar[it] = *(const float4*)(Ain + (size_t)(t0 + row) * K + k0 + c4);
    }
    #pragma unroll
    for (int it = 0; it < 2; ++it) {
      int idx = it * 256 + tid;
      int kk  = idx >> 4;
      int c4  = (idx & 15) << 2;
      wr[it] = *(const float4*)(W + (size_t)(k0 + kk) * HDIM + n0 + c4);
    }
  };

  auto lstore = [&](int b) {
    #pragma unroll
    for (int it = 0; it < 4; ++it) {
      int idx = it * 256 + tid;
      int row = idx >> 3;
      int c4  = (idx & 7) << 2;
      unsigned* dst = (unsigned*)(&Al[b][row * 32 + c4]);
      dst[0] = pk2bf(ar[it].x, ar[it].y);
      dst[1] = pk2bf(ar[it].z, ar[it].w);
    }
    #pragma unroll
    for (int it = 0; it < 2; ++it) {
      int idx = it * 256 + tid;
      int kk  = idx >> 4;
      int c4  = (idx & 15) << 2;
      Bl[b][(c4 + 0) * 32 + kk] = f2bf(wr[it].x);
      Bl[b][(c4 + 1) * 32 + kk] = f2bf(wr[it].y);
      Bl[b][(c4 + 2) * 32 + kk] = f2bf(wr[it].z);
      Bl[b][(c4 + 3) * 32 + kk] = f2bf(wr[it].w);
    }
  };

  const int NK = K / 32;
  gload(0);
  lstore(0);
  __syncthreads();

  #pragma unroll 1
  for (int ks = 0; ks < NK; ++ks) {
    const int cur = ks & 1;
    if (ks + 1 < NK) gload((ks + 1) * 32);

    v16bf af[2], bf2[2];
    #pragma unroll
    for (int mt = 0; mt < 2; ++mt) af[mt] = load_frag(Al[cur], wm * 32 + mt * 16, lane);
    #pragma unroll
    for (int nt = 0; nt < 2; ++nt) bf2[nt] = load_frag(Bl[cur], wn * 32 + nt * 16, lane);
    #pragma unroll
    for (int mt = 0; mt < 2; ++mt)
      #pragma unroll
      for (int nt = 0; nt < 2; ++nt)
        acc[mt][nt] = __builtin_amdgcn_wmma_f32_16x16x32_bf16(
            false, af[mt], false, bf2[nt], (short)0, acc[mt][nt], false, false);

    if (ks + 1 < NK) {
      lstore(cur ^ 1);
      __syncthreads();
    }
  }

  const int rl = (lane >> 4) << 3;
  const int cl = lane & 15;
  #pragma unroll
  for (int mt = 0; mt < 2; ++mt)
    #pragma unroll
    for (int nt = 0; nt < 2; ++nt)
      #pragma unroll
      for (int r = 0; r < 8; ++r) {
        int row = t0 + wm * 32 + mt * 16 + rl + r;
        int col = n0 + wn * 32 + nt * 16 + cl;
        size_t o = (size_t)row * HDIM + col;
        Out[o] += acc[mt][nt][r];
      }
}

// ---------- host ----------

extern "C" void kernel_launch(void* const* d_in, const int* in_sizes, int n_in,
                              void* d_out, int out_size, void* d_ws, size_t ws_size,
                              hipStream_t stream) {
  const float* X   = (const float*)d_in[0];  // [T,H]
  const float* Wr  = (const float*)d_in[1];  // [H,E]
  const float* Wg  = (const float*)d_in[2];  // [E,H,I]
  const float* Wu  = (const float*)d_in[3];  // [E,H,I]
  const float* Wd  = (const float*)d_in[4];  // [E,I,H]
  const float* WsG = (const float*)d_in[5];  // [H,IS]
  const float* WsU = (const float*)d_in[6];  // [H,IS]
  const float* WsD = (const float*)d_in[7];  // [IS,H]
  const float* Wsg = (const float*)d_in[8];  // [H,1]
  float* out = (float*)d_out;

  float* combine = (float*)d_ws;                       // T*E
  float* sg      = combine + (size_t)NTOK * NEXP;      // T
  float* act     = sg + NTOK;                          // up to T*IS

  // 1) router (also zero-inits d_out and computes shared sigmoid gate)
  router_kernel<<<NTOK, 256, 0, stream>>>(X, Wr, Wsg, combine, sg, out);

  // 2) experts: combine weight folded into the gate/up epilogue, down accumulates
  for (int e = 0; e < NEXP; ++e) {
    gateup_kernel<<<dim3(IDIM / 64, NTOK / 128), 256, 0, stream>>>(
        X, Wg + (size_t)e * HDIM * IDIM, Wu + (size_t)e * HDIM * IDIM,
        combine + e, NEXP, act, IDIM);
    down_kernel<<<dim3(HDIM / 64, NTOK / 128), 256, 0, stream>>>(
        act, Wd + (size_t)e * IDIM * HDIM, out, IDIM);
  }

  // 3) shared expert: sigmoid gate folded into gate/up epilogue (stride 1)
  gateup_kernel<<<dim3(ISDIM / 64, NTOK / 128), 256, 0, stream>>>(
      X, WsG, WsU, sg, 1, act, ISDIM);
  down_kernel<<<dim3(HDIM / 64, NTOK / 128), 256, 0, stream>>>(
      act, WsD, out, ISDIM);
}
